// CostVolumeLayer_76587856822669
// MI455X (gfx1250) — matrix operand order repaired and checked by reading, not compiled, not run on hardware
//
#include <hip/hip_runtime.h>

// Cost volume (search range 4) on MI455X via bf16x3 (split-bf16) WMMA.
//
// Roofline: 2.7 GFLOP vs 170 MB compulsory HBM traffic -> 7.3 us floor at
// 23.3 TB/s; needs ~372 TFLOP/s effective compute. bf16 WMMA at 3x
// (hi*hi + hi*lo + lo*hi fp32 emulation, ~16 mantissa bits) clears that.
// Both inputs (128 MB) are L2-resident (192 MB L2); this version processes
// 4 output rows per workgroup so x2 is staged from only 12 rows instead of
// 4x9 = 36, cutting L2 read traffic ~3x (to ~0.4 GB total).
//
// Workgroup = 384 threads = 12 wave32s. 36 (displacement i, row hh) units,
// 3 per wave. Per unit: A = x1[16px x K] of row h0+hh, B = x2 row
// (h0+hh-i)[32px x K] as two 16-column WMMA tiles; the |w-w2| <= 4 band of
// the two 16x16 products covers exactly the 9 horizontal displacements per
// pixel, each output written exactly once.

#define BBATCH 4
#define CC 128
#define HH 128
#define WW 256
#define RSR 4
#define DD 81
#define KC 32           // K chunk = one bf16 WMMA K step
#define HS 4            // output rows per workgroup
#define NROWS 12        // staged x2 rows = HS + 2*RSR
#define BPX 32          // staged x2 pixels per row strip

typedef __attribute__((ext_vector_type(16))) __bf16 v16bf;
typedef __attribute__((ext_vector_type(8)))  float  v8f;
typedef __attribute__((ext_vector_type(4)))  float  vf4;

union Op32 { v16bf v; uint4 q[2]; };   // 32B WMMA operand

// fp32 -> bf16 hi/lo split (truncating): x ~= hi + lo with ~16 mantissa bits.
__device__ __forceinline__ void split2(float x, unsigned& h, unsigned& l) {
  unsigned u = __float_as_uint(x);
  unsigned hh = u >> 16;
  float hf = __uint_as_float(hh << 16);
  float lf = x - hf;
  h = hh;
  l = __float_as_uint(lf) >> 16;
}

__global__ __launch_bounds__(384) void cost_volume_wmma(
    const float* __restrict__ x1, const float* __restrict__ x2,
    float* __restrict__ out)
{
  // LDS staging, WMMA-source layout: per-pixel contiguous K (channel) runs.
  __shared__ __align__(16) unsigned short Ahi[HS * 16 * KC];          // 4 KB
  __shared__ __align__(16) unsigned short Alo[HS * 16 * KC];          // 4 KB
  __shared__ __align__(16) unsigned short Bhi[NROWS * BPX * KC];      // 24 KB
  __shared__ __align__(16) unsigned short Blo[NROWS * BPX * KC];      // 24 KB

  const int t    = threadIdx.x;
  const int w0   = blockIdx.x * 16;    // 16-pixel tile along W
  const int h0   = blockIdx.y * HS;    // 4-row strip
  const int b    = blockIdx.z;

  const int lane = t & 31;
  const int wave = t >> 5;             // 0..11
  const int n    = lane & 15;
  const int half = lane >> 4;          // lane half selects K window

  v8f acc[3][2] = {};                  // 3 (i,hh) units x 2 n-tiles

  // ---- x2 staging assignment: (row, channel-pair, pixel-half) per thread ----
  const int  srr   = t >> 5;           // 0..11 -> x2 row h0-4+srr
  const int  scp   = (t >> 1) & 15;    // channel pair within K chunk
  const int  sph   = (t & 1) * 16;     // pixel half: 0 or 16
  const int  hsrc  = h0 - RSR + srr;
  const bool rowok = (unsigned)hsrc < (unsigned)HH;

  for (int cb = 0; cb < CC; cb += KC) {
    // ---- stage x1 tiles (threads 0..255: 4 rows x 16 ch-pairs x 4 px) ----
    if (t < 256) {
      const int hh  = t >> 6;
      const int rem = t & 63;
      const int cp  = rem >> 2;
      const int q4  = (rem & 3) * 4;
      const size_t base =
          (((size_t)b * CC + cb + 2 * cp) * HH + h0 + hh) * WW + w0 + q4;
      const vf4 v0 = __builtin_nontemporal_load((const vf4*)&x1[base]);
      const vf4 v1 = __builtin_nontemporal_load((const vf4*)&x1[base + (size_t)HH * WW]);
      #pragma unroll
      for (int e = 0; e < 4; ++e) {
        unsigned h0u, l0u, h1u, l1u;
        split2(v0[e], h0u, l0u);
        split2(v1[e], h1u, l1u);
        const int idx = (hh * 16 + q4 + e) * (KC / 2) + cp;   // uint units
        ((unsigned*)Ahi)[idx] = h0u | (h1u << 16);
        ((unsigned*)Alo)[idx] = l0u | (l1u << 16);
      }
    }
    // ---- stage x2 strip (all 384 threads), zero-fill out-of-range ----
    {
      const float* src =
          &x2[(((size_t)b * CC + cb + 2 * scp) * HH + (rowok ? hsrc : 0)) * WW];
      #pragma unroll
      for (int k4 = 0; k4 < 4; ++k4) {
        const int px = sph + k4 * 4;
        const int wc = w0 - RSR + px;            // multiple of 4 -> all-or-none
        vf4 v0 = {0.f, 0.f, 0.f, 0.f};
        vf4 v1 = {0.f, 0.f, 0.f, 0.f};
        if (rowok && (unsigned)wc <= (unsigned)(WW - 4)) {
          v0 = *(const vf4*)&src[wc];                       // L2-resident
          v1 = *(const vf4*)&src[(size_t)HH * WW + wc];
        }
        #pragma unroll
        for (int e = 0; e < 4; ++e) {
          unsigned h0u, l0u, h1u, l1u;
          split2(v0[e], h0u, l0u);
          split2(v1[e], h1u, l1u);
          const int idx = (srr * BPX + px + e) * (KC / 2) + scp;  // uint units
          ((unsigned*)Bhi)[idx] = h0u | (h1u << 16);
          ((unsigned*)Blo)[idx] = l0u | (l1u << 16);
        }
      }
    }
    __syncthreads();

    // ---- WMMA: wave handles units u, u+12, u+24 of 36 (i_idx, hh) units ----
    #pragma unroll
    for (int p = 0; p < 3; ++p) {
      const int unit = wave + 12 * p;
      const int iidx = unit >> 2;      // 0..8 -> displacement i = iidx-4
      const int hh   = unit & 3;
      const int rr   = hh + 8 - iidx;  // staged row for x2[h0+hh - i]

      // A operand, ISA 16-bit A 16x32 layout:
      //   lanes 0-15: K=0..7 (V0-3), K=16..23 (V4-7); lanes 16-31: +8.
      Op32 ahi, alo;
      const int abase = (hh * 16 + n) * KC + half * 8;
      ahi.q[0] = *(const uint4*)&Ahi[abase];
      ahi.q[1] = *(const uint4*)&Ahi[abase + 16];
      alo.q[0] = *(const uint4*)&Alo[abase];
      alo.q[1] = *(const uint4*)&Alo[abase + 16];

      #pragma unroll
      for (int nt = 0; nt < 2; ++nt) {
        // B 32x16 layout: lane holds column (pixel); lanes 0-15 K=0..7,16..23,
        // lanes 16-31 K=8..15,24..31 (same K order as A).
        const int bbase = (rr * BPX + nt * 16 + n) * KC + half * 8;
        Op32 bhi, blo;
        const uint4* pbh = (const uint4*)&Bhi[bbase];
        const uint4* pbl = (const uint4*)&Blo[bbase];
        bhi.q[0] = pbh[0]; bhi.q[1] = pbh[2];
        blo.q[0] = pbl[0]; blo.q[1] = pbl[2];

        v8f a = acc[p][nt];
        a = __builtin_amdgcn_wmma_f32_16x16x32_bf16(
            false, ahi.v, false, bhi.v, (short)0, a, false, false);
        a = __builtin_amdgcn_wmma_f32_16x16x32_bf16(
            false, ahi.v, false, blo.v, (short)0, a, false, false);
        a = __builtin_amdgcn_wmma_f32_16x16x32_bf16(
            false, alo.v, false, bhi.v, (short)0, a, false, false);
        acc[p][nt] = a;
      }
    }
    __syncthreads();
  }

  // ---- epilogue: scatter the |j|<=4 band; each output written exactly once.
  // C/D layout: lanes 0-15 N=lane, VGPR v -> M=v; lanes 16-31 N=lane-16, M=8+v.
  const float scale = 1.0f / (float)DD;
  #pragma unroll
  for (int p = 0; p < 3; ++p) {
    const int unit  = wave + 12 * p;
    const int idisp = (unit >> 2) - 4;
    const int h     = h0 + (unit & 3);
    #pragma unroll
    for (int v = 0; v < 8; ++v) {
      const int m = v + half * 8;
      const int w = w0 + m;
      {
        const int j = m + 4 - n;             // n-tile 0: w2 = w0-4+n
        if (j >= -RSR && j <= RSR) {
          const int k = (9 * idisp + j + DD) % DD;
          __builtin_nontemporal_store(
              acc[p][0][v] * scale,
              &out[(((size_t)b * DD + k) * HH + h) * WW + w]);
        }
      }
      {
        const int j = m + 4 - 16 - n;        // n-tile 1: w2 = w0+12+n
        if (j >= -RSR && j <= RSR) {
          const int k = (9 * idisp + j + DD) % DD;
          __builtin_nontemporal_store(
              acc[p][1][v] * scale,
              &out[(((size_t)b * DD + k) * HH + h) * WW + w]);
        }
      }
    }
  }
}

extern "C" void kernel_launch(void* const* d_in, const int* in_sizes, int n_in,
                              void* d_out, int out_size, void* d_ws, size_t ws_size,
                              hipStream_t stream) {
  (void)in_sizes; (void)n_in; (void)out_size; (void)d_ws; (void)ws_size;
  const float* x1 = (const float*)d_in[0];
  const float* x2 = (const float*)d_in[1];
  float* out = (float*)d_out;
  dim3 grid(WW / 16, HH / HS, BBATCH);   // 16 x 32 x 4 = 2048 workgroups
  cost_volume_wmma<<<grid, dim3(384), 0, stream>>>(x1, x2, out);
}
